// GAT_72791105733218
// MI455X (gfx1250) — matrix operand order repaired and checked by reading, not compiled, not run on hardware
//
#include <hip/hip_runtime.h>
#include <hip/hip_fp16.h>
#include <math.h>

typedef __attribute__((ext_vector_type(16))) _Float16 v16h;
typedef __attribute__((ext_vector_type(8)))  _Float16 v8h;
typedef __attribute__((ext_vector_type(8)))  float    v8f;

#define NF    602   // input features
#define H1    8     // heads layer 1
#define C1    8     // hidden per head
#define F1OUT 64    // H1*C1
#define NC    41    // classes (layer 2, 1 head)
#define NEG_SLOPE 0.2f
#define EPS 1e-16f

// ---------------------------------------------------------------------------
// Init: zero aggregates / denominators, -inf the segment-max buffers.
// ---------------------------------------------------------------------------
__global__ void init_kernel(float* m1, float* d1, float* agg1,
                            float* m2, float* d2, float* agg2, int N) {
  int i = blockIdx.x * blockDim.x + threadIdx.x;
  if (i < N * F1OUT) agg1[i] = 0.f;
  if (i < N * NC)    agg2[i] = 0.f;
  if (i < N * H1)  { m1[i] = -__builtin_inff(); d1[i] = 0.f; }
  if (i < N)       { m2[i] = -__builtin_inff(); d2[i] = 0.f; }
}

// ---------------------------------------------------------------------------
// WMMA GEMM: C[M,NCOLS] = A[M,K] (f32 row-major) * B[K,NCOLS] (f32 row-major)
// fp16 multiplicands, fp32 accumulate via v_wmma_f32_16x16x32_f16.
//
// Block = 128 threads = 4 wave32; wave owns a 16-row strip x NT 16-col tiles.
// B is staged once into LDS (transposed [col][k], f16, zero-padded in K and
// cols) so the hot loop does: vector global loads (A) + 2x ds_load_b128 per
// B fragment + NT wmma per 32-wide K chunk — no guards, EXEC stays all-ones.
// A stream is HBM-bound (23.3 TB/s roofline), so prefetch one chunk ahead.
//
// Fragment layouts per CDNA5 ISA 7.12.2 (wave32):
//   A 16x32 f16 : lane l(0..15)=row, half=lane>>4 ; elem i<8 -> K=half*8+i,
//                 elem i>=8 -> K=16+half*8+(i-8)
//   B 32x16 f16 : lane l(0..15)=col ; elem i<8 -> K=half*8+i, elem i>=8 -> K=16+half*8+(i-8)
//   C 16x16 f32 : vgpr r : row = half*8 + r, col = lane&15
// ---------------------------------------------------------------------------
template <int K, int NCOLS, int NT>
__global__ __launch_bounds__(128)
void wmma_gemm(const float* __restrict__ A, const float* __restrict__ B,
               float* __restrict__ C, int M) {
  constexpr int KCH   = (K + 31) / 32;   // chunks incl. tail
  constexpr int KMAIN = K / 32;          // full chunks
  constexpr int KPAD  = KCH * 32;
  constexpr int LCOLS = NT * 16;
  __shared__ _Float16 Bs[LCOLS * KPAD];

  // ---- cooperative B -> LDS (transpose, cvt f16, zero-pad K and cols) ----
  for (int idx = threadIdx.x; idx < LCOLS * KPAD; idx += 128) {
    const int col = idx / KPAD;
    const int k   = idx % KPAD;
    const float v = (col < NCOLS && k < K) ? B[(size_t)k * NCOLS + col] : 0.f;
    Bs[col * KPAD + k] = (_Float16)v;
  }
  __syncthreads();

  const int lane = threadIdx.x & 31;
  const int wave = threadIdx.x >> 5;
  const int half = lane >> 4;
  const int l    = lane & 15;
  const int rowBase = blockIdx.x * 64 + wave * 16;
  int arow = rowBase + l;
  if (arow > M - 1) arow = M - 1;  // clamp: OOB rows compute on row M-1, discarded at store
  const float* Ap = A + (size_t)arow * K;

  v8f acc[NT] = {};

  // ---- main loop: unconditional vector loads ----
  for (int c = 0; c < KMAIN; ++c) {
    const int k0 = c * 32;
    const int ka = k0 + half * 8;
    // speculative prefetch one K-chunk ahead (global_prefetch_b8; OOB is
    // silently dropped per ISA 10.5, no counters touched)
    __builtin_prefetch(Ap + ka + 32, 0, 0);
    v16h a;
    const float2* A2a = (const float2*)(Ap + ka);        // 8B-aligned
    const float2* A2b = (const float2*)(Ap + ka + 16);
#pragma unroll
    for (int j = 0; j < 4; ++j) {
      const float2 p = A2a[j];
      const float2 q = A2b[j];
      a[2 * j]     = (_Float16)p.x;  a[2 * j + 1]     = (_Float16)p.y;
      a[8 + 2 * j] = (_Float16)q.x;  a[8 + 2 * j + 1] = (_Float16)q.y;
    }
    const int kb = k0 + half * 8;
#pragma unroll
    for (int t = 0; t < NT; ++t) {
      const _Float16* bp = Bs + (t * 16 + l) * KPAD + kb;  // 16B-aligned
      const v8h b0 = *(const v8h*)(bp);
      const v8h b1 = *(const v8h*)(bp + 16);
      v16h b;
#pragma unroll
      for (int i = 0; i < 8; ++i) { b[i] = b0[i]; b[i + 8] = b1[i]; }
      acc[t] = __builtin_amdgcn_wmma_f32_16x16x32_f16(
          false, a, false, b, (short)0, acc[t], false, false);
    }
  }

  // ---- K tail (clamped unconditional loads + value selects) ----
  if constexpr (K % 32 != 0) {
    const int k0 = KMAIN * 32;
    const int ka = k0 + half * 8;
    v16h a;
#pragma unroll
    for (int i = 0; i < 8; ++i) {
      const int k1 = ka + i, k2 = ka + 16 + i;
      const float va = Ap[k1 < K ? k1 : K - 1];
      const float vb = Ap[k2 < K ? k2 : K - 1];
      a[i]     = (k1 < K) ? (_Float16)va : (_Float16)0.f;
      a[i + 8] = (k2 < K) ? (_Float16)vb : (_Float16)0.f;
    }
    const int kb = k0 + half * 8;
#pragma unroll
    for (int t = 0; t < NT; ++t) {
      const _Float16* bp = Bs + (t * 16 + l) * KPAD + kb;  // zero-padded
      const v8h b0 = *(const v8h*)(bp);
      const v8h b1 = *(const v8h*)(bp + 16);
      v16h b;
#pragma unroll
      for (int i = 0; i < 8; ++i) { b[i] = b0[i]; b[i + 8] = b1[i]; }
      acc[t] = __builtin_amdgcn_wmma_f32_16x16x32_f16(
          false, a, false, b, (short)0, acc[t], false, false);
    }
  }

  // ---- store C ----
#pragma unroll
  for (int t = 0; t < NT; ++t) {
    const int col = t * 16 + l;
    if (col >= NCOLS) continue;
#pragma unroll
    for (int r = 0; r < 8; ++r) {
      const int row = rowBase + half * 8 + r;
      if (row < M) C[(size_t)row * NCOLS + col] = acc[t][r];
    }
  }
}

// ---------------------------------------------------------------------------
// Per-node attention logits, layer 1:  a_src[n,h] = <h1[n,h,:], att_src[h,:]>
// h1 head-rows are 32B and 16B-aligned -> float4 loads.
// ---------------------------------------------------------------------------
__global__ void att1_kernel(const float* __restrict__ h1,
                            const float* __restrict__ as,
                            const float* __restrict__ ad,
                            float* a_src, float* a_dst, int N) {
  int i = blockIdx.x * blockDim.x + threadIdx.x;
  if (i >= N * H1) return;
  const int n = i >> 3, h = i & 7;
  const float4* row = (const float4*)(h1 + (size_t)n * F1OUT + h * C1);
  const float4* asv = (const float4*)(as + h * C1);
  const float4* adv = (const float4*)(ad + h * C1);
  float s = 0.f, d = 0.f;
#pragma unroll
  for (int j = 0; j < 2; ++j) {
    const float4 v = row[j], a4 = asv[j], d4 = adv[j];
    s += v.x * a4.x + v.y * a4.y + v.z * a4.z + v.w * a4.w;
    d += v.x * d4.x + v.y * d4.y + v.z * d4.z + v.w * d4.w;
  }
  a_src[i] = s;
  a_dst[i] = d;
}

__device__ __forceinline__ void load8(const float* __restrict__ p, float* v) {
  const float4 a = ((const float4*)p)[0];
  const float4 b = ((const float4*)p)[1];
  v[0] = a.x; v[1] = a.y; v[2] = a.z; v[3] = a.w;
  v[4] = b.x; v[5] = b.y; v[6] = b.z; v[7] = b.w;
}

// ---------------------------------------------------------------------------
// Edge pass 1: segment max of leaky_relu(a_src[src]+a_dst[dst])  (layer 1)
// ---------------------------------------------------------------------------
__global__ void edge_max1(const int* __restrict__ ei,
                          const float* __restrict__ a_src,
                          const float* __restrict__ a_dst,
                          float* m1, int E) {
  int e = blockIdx.x * blockDim.x + threadIdx.x;
  if (e >= E) return;
  const int s = ei[e], d = ei[E + e];
  float vs[H1], vd[H1];
  load8(a_src + (size_t)s * H1, vs);   // 2x global_load_b128
  load8(a_dst + (size_t)d * H1, vd);
#pragma unroll
  for (int h = 0; h < H1; ++h) {
    float v = vs[h] + vd[h];
    v = v > 0.f ? v : NEG_SLOPE * v;
    atomicMax(&m1[d * H1 + h], v);     // global_atomic_max_num_f32
  }
}

// Edge pass 2: w = exp(e - max); segment-sum denominators  (layer 1)
__global__ void edge_exp1(const int* __restrict__ ei,
                          const float* __restrict__ a_src,
                          const float* __restrict__ a_dst,
                          const float* __restrict__ m1,
                          float* __restrict__ w, float* d1, int E) {
  int e = blockIdx.x * blockDim.x + threadIdx.x;
  if (e >= E) return;
  const int s = ei[e], d = ei[E + e];
  float vs[H1], vd[H1], vm[H1];
  load8(a_src + (size_t)s * H1, vs);
  load8(a_dst + (size_t)d * H1, vd);
  load8(m1 + (size_t)d * H1, vm);
#pragma unroll
  for (int h = 0; h < H1; ++h) {
    float v = vs[h] + vd[h];
    v = v > 0.f ? v : NEG_SLOPE * v;
    const float wv = expf(v - vm[h]);
    w[(size_t)e * H1 + h] = wv;
    atomicAdd(&d1[d * H1 + h], wv);
  }
}

// Edge pass 3: out[dst] += alpha * h1[src]  (layer 1, thread per edge*head)
__global__ void edge_agg1(const int* __restrict__ ei,
                          const float* __restrict__ h1,
                          const float* __restrict__ w,
                          const float* __restrict__ d1,
                          float* agg, int E) {
  int i = blockIdx.x * blockDim.x + threadIdx.x;
  if (i >= E * H1) return;
  const int e = i >> 3, h = i & 7;
  const int s = ei[e], d = ei[E + e];
  const float alpha = w[i] / (d1[d * H1 + h] + EPS);
  float hs[C1];
  load8(h1 + (size_t)s * F1OUT + h * C1, hs);  // 2x global_load_b128 gather
  float* od = agg + (size_t)d * F1OUT + h * C1;
#pragma unroll
  for (int c = 0; c < C1; ++c) atomicAdd(&od[c], alpha * hs[c]);
}

// f1 = elu(agg1 + b1)
__global__ void elu1_kernel(const float* __restrict__ agg,
                            const float* __restrict__ b1, float* f1, int N) {
  int i = blockIdx.x * blockDim.x + threadIdx.x;
  if (i >= N * F1OUT) return;
  const float v = agg[i] + b1[i % F1OUT];
  f1[i] = v > 0.f ? v : expm1f(v);
}

// Per-node attention logits, layer 2 (1 head, 41 channels)
__global__ void att2_kernel(const float* __restrict__ h2,
                            const float* __restrict__ as,
                            const float* __restrict__ ad,
                            float* a_src, float* a_dst, int N) {
  int n = blockIdx.x * blockDim.x + threadIdx.x;
  if (n >= N) return;
  const float* row = h2 + (size_t)n * NC;
  float s = 0.f, d = 0.f;
  for (int c = 0; c < NC; ++c) {
    const float v = row[c];
    s += v * as[c];
    d += v * ad[c];
  }
  a_src[n] = s;
  a_dst[n] = d;
}

__global__ void edge_max2(const int* __restrict__ ei,
                          const float* __restrict__ a_src,
                          const float* __restrict__ a_dst, float* m2, int E) {
  int e = blockIdx.x * blockDim.x + threadIdx.x;
  if (e >= E) return;
  const int s = ei[e], d = ei[E + e];
  float v = a_src[s] + a_dst[d];
  v = v > 0.f ? v : NEG_SLOPE * v;
  atomicMax(&m2[d], v);
}

__global__ void edge_exp2(const int* __restrict__ ei,
                          const float* __restrict__ a_src,
                          const float* __restrict__ a_dst,
                          const float* __restrict__ m2,
                          float* __restrict__ w, float* d2, int E) {
  int e = blockIdx.x * blockDim.x + threadIdx.x;
  if (e >= E) return;
  const int s = ei[e], d = ei[E + e];
  float v = a_src[s] + a_dst[d];
  v = v > 0.f ? v : NEG_SLOPE * v;
  const float wv = expf(v - m2[d]);
  w[e] = wv;
  atomicAdd(&d2[d], wv);
}

__global__ void edge_agg2(const int* __restrict__ ei,
                          const float* __restrict__ h2,
                          const float* __restrict__ w,
                          const float* __restrict__ d2,
                          float* agg, int E) {
  int e = blockIdx.x * blockDim.x + threadIdx.x;
  if (e >= E) return;
  const int s = ei[e], d = ei[E + e];
  const float alpha = w[e] / (d2[d] + EPS);
  const float* hs = h2 + (size_t)s * NC;
  float* od = agg + (size_t)d * NC;
  for (int c = 0; c < NC; ++c) atomicAdd(&od[c], alpha * hs[c]);
}

// out = log_softmax(agg2 + b2) over 41 classes (1 head -> mean is identity)
__global__ void lsm_kernel(const float* __restrict__ agg,
                           const float* __restrict__ b2, float* out, int N) {
  int n = blockIdx.x * blockDim.x + threadIdx.x;
  if (n >= N) return;
  float v[NC];
  float mx = -__builtin_inff();
  for (int c = 0; c < NC; ++c) {
    v[c] = agg[(size_t)n * NC + c] + b2[c];
    mx = fmaxf(mx, v[c]);
  }
  float s = 0.f;
  for (int c = 0; c < NC; ++c) s += expf(v[c] - mx);
  const float ls = logf(s);
  for (int c = 0; c < NC; ++c) out[(size_t)n * NC + c] = v[c] - mx - ls;
}

// ---------------------------------------------------------------------------
extern "C" void kernel_launch(void* const* d_in, const int* in_sizes, int n_in,
                              void* d_out, int out_size, void* d_ws, size_t ws_size,
                              hipStream_t stream) {
  (void)n_in; (void)out_size; (void)ws_size;
  const float* x   = (const float*)d_in[0];
  const int*   ei  = (const int*)d_in[1];   // [2,E]: row0=src, row1=dst
  const float* W1  = (const float*)d_in[2];
  const float* as1 = (const float*)d_in[3];
  const float* ad1 = (const float*)d_in[4];
  const float* b1  = (const float*)d_in[5];
  const float* W2  = (const float*)d_in[6];
  const float* as2 = (const float*)d_in[7];
  const float* ad2 = (const float*)d_in[8];
  const float* b2  = (const float*)d_in[9];
  float* out = (float*)d_out;

  const int N = in_sizes[0] / NF;
  const int E = in_sizes[1] / 2;

  // ---- carve workspace ----
  char* ws = (char*)d_ws;
  size_t off = 0;
  auto alloc = [&](size_t bytes) -> float* {
    float* p = (float*)(ws + off);
    off = (off + bytes + 255) & ~(size_t)255;
    return p;
  };
  float* h1   = alloc((size_t)N * F1OUT * 4);
  float* asr1 = alloc((size_t)N * H1 * 4);
  float* adt1 = alloc((size_t)N * H1 * 4);
  float* m1   = alloc((size_t)N * H1 * 4);
  float* dn1  = alloc((size_t)N * H1 * 4);
  float* w1e  = alloc((size_t)E * H1 * 4);
  float* agg1 = alloc((size_t)N * F1OUT * 4);
  float* f1   = alloc((size_t)N * F1OUT * 4);
  float* h2   = alloc((size_t)N * NC * 4);
  float* asr2 = alloc((size_t)N * 4);
  float* adt2 = alloc((size_t)N * 4);
  float* m2   = alloc((size_t)N * 4);
  float* dn2  = alloc((size_t)N * 4);
  float* w2e  = alloc((size_t)E * 4);
  float* agg2 = alloc((size_t)N * NC * 4);

  const int T = 256;
  auto cdiv = [](long long a, long long b) { return (int)((a + b - 1) / b); };

  init_kernel<<<cdiv((long long)N * F1OUT, T), T, 0, stream>>>(m1, dn1, agg1, m2, dn2, agg2, N);

  // ---- layer 1 ----
  wmma_gemm<NF, F1OUT, 4><<<cdiv(N, 64), 128, 0, stream>>>(x, W1, h1, N);
  att1_kernel<<<cdiv((long long)N * H1, T), T, 0, stream>>>(h1, as1, ad1, asr1, adt1, N);
  edge_max1<<<cdiv(E, T), T, 0, stream>>>(ei, asr1, adt1, m1, E);
  edge_exp1<<<cdiv(E, T), T, 0, stream>>>(ei, asr1, adt1, m1, w1e, dn1, E);
  edge_agg1<<<cdiv((long long)E * H1, T), T, 0, stream>>>(ei, h1, w1e, dn1, agg1, E);
  elu1_kernel<<<cdiv((long long)N * F1OUT, T), T, 0, stream>>>(agg1, b1, f1, N);

  // ---- layer 2 ----
  wmma_gemm<F1OUT, NC, 3><<<cdiv(N, 64), 128, 0, stream>>>(f1, W2, h2, N);
  att2_kernel<<<cdiv(N, T), T, 0, stream>>>(h2, as2, ad2, asr2, adt2, N);
  edge_max2<<<cdiv(E, T), T, 0, stream>>>(ei, asr2, adt2, m2, E);
  edge_exp2<<<cdiv(E, T), T, 0, stream>>>(ei, asr2, adt2, m2, w2e, dn2, E);
  edge_agg2<<<cdiv(E, T), T, 0, stream>>>(ei, h2, w2e, dn2, agg2, E);
  lsm_kernel<<<cdiv(N, T), T, 0, stream>>>(agg2, b2, out, N);
}
